// GraphMoeV21ConfHier_72267119722664
// MI455X (gfx1250) — matrix operand order repaired and checked by reading, not compiled, not run on hardware
//
#include <hip/hip_runtime.h>

#define N_NODES 20000
#define N_EDGES 320000
#define D 256
#define R 512          // 2*D router/expert input
#define NG 4
#define NE 8
#define OUTD 128
#define NL 3

typedef _Float16 half_t;
typedef __attribute__((ext_vector_type(16))) _Float16 v16h;
typedef __attribute__((ext_vector_type(8)))  float    v8f;

// ---------------- WMMA fragment helpers (CDNA5 ISA 7.12.2 layouts) ----------

// A: 16x32 tile fragment. A stored row-major in LDS, ld in halfs.
// lanes 0-15: row=lane, VGPR0..3 -> K=0..7, VGPR4..7 -> K=16..23
// lanes16-31: row=lane-16, VGPR0..3 -> K=8..15, VGPR4..7 -> K=24..31
// Per lane: two contiguous 8-half (16B) runs -> compiler emits 2x ds_load_b128.
__device__ __forceinline__ v16h load_a_frag(const half_t* A, int ld, int lane) {
  const int row  = lane & 15;
  const int hsel = lane >> 4;
  v16h a;
#pragma unroll
  for (int j = 0; j < 8; ++j) {
    const int kb = ((j < 4) ? 0 : 16) + hsel * 8 + 2 * (j & 3);
    a[2 * j]     = A[row * ld + kb];
    a[2 * j + 1] = A[row * ld + kb + 1];
  }
  return a;
}

// B: weights pre-packed fragment-ready. For tile (kt,nt) lane l, the 16 halfs
// of the lane's fragment are contiguous: one 32-byte load (2x global_load_b128).
__device__ __forceinline__ v16h load_b_packed(const half_t* __restrict__ Bp,
                                              int kt, int nt, int ntiles, int lane) {
  return *(const v16h*)(Bp + ((((size_t)kt * ntiles + nt) * 32) + lane) * 16);
}

__device__ __forceinline__ v8f wmma16(v16h a, v16h b, v8f c) {
  return __builtin_amdgcn_wmma_f32_16x16x32_f16(false, a, false, b, (short)0, c,
                                                false, false);
}

// D/C element mapping: value j of v8f -> row m = j + 8*(lane>>4), col n = lane&15.

// ---------------- utility kernels ------------------------------------------

__global__ void k_zero_f32(float* p, int n) {
  int i = blockIdx.x * blockDim.x + threadIdx.x;
  if (i < n) p[i] = 0.0f;
}
__global__ void k_zero_i32(int* p, int n) {
  int i = blockIdx.x * blockDim.x + threadIdx.x;
  if (i < n) p[i] = 0;
}
__global__ void k_f32_to_f16(const float* in, half_t* out, int n) {
  int i = blockIdx.x * blockDim.x + threadIdx.x;
  if (i < n) out[i] = (half_t)in[i];
}

// Pack f32 weights [nmat, K, N] into fragment-ready f16 layout.
// out element index: (((mat*K/32 + kt)*(N/16) + nt)*32 + lane)*16 + v
// maps to W[mat][kt*32 + (v>>3)*16 + (lane>>4)*8 + (v&7)][nt*16 + (lane&15)]
__global__ void k_pack_b(const float* __restrict__ W, half_t* __restrict__ out,
                         int K, int N, int nmat) {
  long idx   = (long)blockIdx.x * blockDim.x + threadIdx.x;
  long per   = (long)K * N;
  long total = (long)nmat * per;
  if (idx >= total) return;
  int  mat = (int)(idx / per);
  long rr  = idx - (long)mat * per;
  int  v    = (int)(rr & 15);
  int  l    = (int)((rr >> 4) & 31);
  long tile = rr >> 9;
  int  ntiles = N >> 4;
  int  nt = (int)(tile % ntiles);
  int  kt = (int)(tile / ntiles);
  int  k   = kt * 32 + ((v >> 3) * 16) + ((l >> 4) * 8) + (v & 7);
  int  col = nt * 16 + (l & 15);
  out[idx] = (half_t)W[(size_t)mat * per + (size_t)k * N + col];
}

// ---------------- graph kernels --------------------------------------------

__global__ void k_degree(const int* dst, float* deg) {
  int e = blockIdx.x * blockDim.x + threadIdx.x;
  if (e < N_EDGES) atomicAdd(&deg[dst[e]], 1.0f);
}

// one wave (32 lanes) per edge; each lane moves 8 floats
__global__ void k_scatter(const float* __restrict__ h, const int* __restrict__ src,
                          const int* __restrict__ dst, float* __restrict__ agg) {
  int gid  = blockIdx.x * blockDim.x + threadIdx.x;
  int e    = gid >> 5;
  int lane = gid & 31;
  if (e >= N_EDGES) return;
  int s = src[e], d = dst[e];
  const float4* hs = (const float4*)(h + (size_t)s * D);
  float4 a = hs[lane * 2 + 0];
  float4 b = hs[lane * 2 + 1];
  float* o = agg + (size_t)d * D + lane * 8;
  atomicAdd(o + 0, a.x); atomicAdd(o + 1, a.y);
  atomicAdd(o + 2, a.z); atomicAdd(o + 3, a.w);
  atomicAdd(o + 4, b.x); atomicAdd(o + 5, b.y);
  atomicAdd(o + 6, b.z); atomicAdd(o + 7, b.w);
}

// ---------------- router: one wave per node --------------------------------

__global__ void k_router(const float* __restrict__ h, const float* __restrict__ agg,
                         const float* __restrict__ deg, const float* __restrict__ Wg,
                         const float* __restrict__ We, const float* __restrict__ wc,
                         half_t* __restrict__ r16, float* __restrict__ conf,
                         int* __restrict__ topi, float* __restrict__ topg,
                         int* __restrict__ counts) {
  int node = blockIdx.x * (blockDim.x >> 5) + (threadIdx.x >> 5);
  int lane = threadIdx.x & 31;
  if (node >= N_NODES) return;
  float dg  = deg[node];
  float inv = 1.0f / (dg < 1.0f ? 1.0f : dg);
  float acc[13];
#pragma unroll
  for (int j = 0; j < 13; ++j) acc[j] = 0.0f;
#pragma unroll
  for (int i = 0; i < 16; ++i) {
    int   k = lane + 32 * i;
    float v = (k < D) ? h[(size_t)node * D + k] : agg[(size_t)node * D + (k - D)] * inv;
    r16[(size_t)node * R + k] = (half_t)v;
#pragma unroll
    for (int g = 0; g < NG; ++g) acc[g] += v * Wg[k * NG + g];
#pragma unroll
    for (int e = 0; e < NE; ++e) acc[4 + e] += v * We[k * NE + e];
    acc[12] += v * wc[k];
  }
#pragma unroll
  for (int off = 16; off > 0; off >>= 1) {
#pragma unroll
    for (int j = 0; j < 13; ++j) acc[j] += __shfl_down(acc[j], off, 32);
  }
  if (lane == 0) {
    float m = acc[0];
    for (int g = 1; g < NG; ++g) m = fmaxf(m, acc[g]);
    float gp[NG], s = 0.0f;
    for (int g = 0; g < NG; ++g) { gp[g] = __expf(acc[g] - m); s += gp[g]; }
    float probs[NE];
    for (int g = 0; g < NG; ++g) {
      float a0 = acc[4 + 2 * g], a1 = acc[4 + 2 * g + 1];
      float mm = fmaxf(a0, a1);
      float e0 = __expf(a0 - mm), e1 = __expf(a1 - mm);
      float gw = gp[g] / (s * (e0 + e1));
      probs[2 * g]     = gw * e0;
      probs[2 * g + 1] = gw * e1;
    }
    int i0 = 0;
    for (int e = 1; e < NE; ++e) if (probs[e] > probs[i0]) i0 = e;
    int i1 = (i0 == 0) ? 1 : 0;
    for (int e = 0; e < NE; ++e)
      if (e != i0 && probs[e] > probs[i1]) i1 = e;
    float v0 = probs[i0], v1 = probs[i1];
    float rn = 1.0f / (v0 + v1 + 1e-9f);
    topi[2 * node]     = i0;  topi[2 * node + 1] = i1;
    topg[2 * node]     = v0 * rn; topg[2 * node + 1] = v1 * rn;
    conf[node] = 1.0f / (1.0f + __expf(-acc[12]));
    atomicAdd(&counts[i0], 1);
    atomicAdd(&counts[i1], 1);
  }
}

__global__ void k_offsets(const int* counts, int* offsets, int* cursors) {
  if (threadIdx.x == 0 && blockIdx.x == 0) {
    int s = 0;
    for (int e = 0; e < NE; ++e) { offsets[e] = s; cursors[e] = s; s += counts[e]; }
    offsets[NE] = s;
  }
}

__global__ void k_assign(const int* __restrict__ topi, const float* __restrict__ topg,
                         int* __restrict__ cursors, int* __restrict__ asg_node,
                         float* __restrict__ asg_gate) {
  int n = blockIdx.x * blockDim.x + threadIdx.x;
  if (n >= N_NODES) return;
#pragma unroll
  for (int k = 0; k < 2; ++k) {
    int e   = topi[2 * n + k];
    int pos = atomicAdd(&cursors[e], 1);
    asg_node[pos] = n;
    asg_gate[pos] = topg[2 * n + k];
  }
}

// ---------------- expert MLP: 16-row tile per block, WMMA ------------------
// GEMM1: [16,512] x W1[e][512,256] -> relu -> GEMM2: [16,256] x W2[e][256,256]
// 128 threads = 4 waves; each wave owns 4 column tiles of 16.

__global__ void __launch_bounds__(128) k_expert(
    const half_t* __restrict__ r16, const half_t* __restrict__ W1p,
    const half_t* __restrict__ W2p, const float* __restrict__ b1,
    const float* __restrict__ b2, const int* __restrict__ offsets,
    const int* __restrict__ asg_node, const float* __restrict__ asg_gate,
    float* __restrict__ strong) {
  __shared__ half_t ldsA[16 * R];  // 16 KB
  __shared__ half_t ldsH[16 * D];  // 8 KB

  // map block -> (expert, row tile)
  int b = blockIdx.x, e = -1, rowBase = 0, accT = 0;
  for (int i = 0; i < NE; ++i) {
    int cnt = offsets[i + 1] - offsets[i];
    int t   = (cnt + 15) >> 4;
    if (b < accT + t) { e = i; rowBase = offsets[i] + (b - accT) * 16; break; }
    accT += t;
  }
  if (e < 0) return;  // uniform per block; EXEC stays full for WMMA
  int rowEnd = offsets[e + 1];
  int tid = threadIdx.x, wave = tid >> 5, lane = tid & 31;

  // stage A tile (zero-padded rows)
  for (int i = tid; i < 16 * R; i += 128) {
    int m = i >> 9, k = i & (R - 1);
    int a = rowBase + m;
    half_t v = (half_t)0.0f;
    if (a < rowEnd) v = r16[(size_t)asg_node[a] * R + k];
    ldsA[i] = v;
  }
  __syncthreads();

  const half_t* W1e = W1p + (size_t)e * R * D;  // packed
  v8f zero = {};
  v8f acc1[4] = {zero, zero, zero, zero};
  for (int ks = 0; ks < R; ks += 32) {
    v16h af = load_a_frag(ldsA + ks, R, lane);
    int kt = ks >> 5;
#pragma unroll
    for (int c = 0; c < 4; ++c) {
      v16h bf = load_b_packed(W1e, kt, wave * 4 + c, D / 16, lane);
      acc1[c] = wmma16(af, bf, acc1[c]);
    }
  }
  // hid = relu(acc + b1) -> LDS (f16)
  const float* b1e = b1 + (size_t)e * D;
#pragma unroll
  for (int c = 0; c < 4; ++c) {
    int colT = (wave * 4 + c) * 16;
#pragma unroll
    for (int j = 0; j < 8; ++j) {
      int m = j + ((lane >> 4) << 3);
      int n = (lane & 15) + colT;
      float v = acc1[c][j] + b1e[n];
      ldsH[m * D + n] = (half_t)(v > 0.0f ? v : 0.0f);
    }
  }
  __syncthreads();

  const half_t* W2e = W2p + (size_t)e * D * D;  // packed
  v8f acc2[4] = {zero, zero, zero, zero};
  for (int ks = 0; ks < D; ks += 32) {
    v16h af = load_a_frag(ldsH + ks, D, lane);
    int kt = ks >> 5;
#pragma unroll
    for (int c = 0; c < 4; ++c) {
      v16h bf = load_b_packed(W2e, kt, wave * 4 + c, D / 16, lane);
      acc2[c] = wmma16(af, bf, acc2[c]);
    }
  }
  // gated scatter into strong
  const float* b2e = b2 + (size_t)e * D;
#pragma unroll
  for (int c = 0; c < 4; ++c) {
    int colT = (wave * 4 + c) * 16;
#pragma unroll
    for (int j = 0; j < 8; ++j) {
      int m = j + ((lane >> 4) << 3);
      int n = (lane & 15) + colT;
      int a = rowBase + m;
      if (a < rowEnd) {
        int node = asg_node[a];
        float g  = asg_gate[a];
        atomicAdd(&strong[(size_t)node * D + n], g * (acc2[c][j] + b2e[n]));
      }
    }
  }
}

// ---------------- dense GEMM C[n,256] = A16[n,512] @ Bp[512,256] (weak) ----

__global__ void __launch_bounds__(128) k_weak_gemm(
    const half_t* __restrict__ A16, const half_t* __restrict__ Bp,
    float* __restrict__ Cout, int nrows) {
  __shared__ half_t ldsA[16 * R];
  int row0 = blockIdx.x * 16;
  int tid = threadIdx.x, wave = tid >> 5, lane = tid & 31;
  for (int i = tid; i < 16 * R; i += 128) {
    int m = i >> 9, k = i & (R - 1);
    int rr = row0 + m;
    ldsA[i] = (rr < nrows) ? A16[(size_t)rr * R + k] : (half_t)0.0f;
  }
  __syncthreads();
  v8f zero = {};
  v8f acc[4] = {zero, zero, zero, zero};
  for (int ks = 0; ks < R; ks += 32) {
    v16h af = load_a_frag(ldsA + ks, R, lane);
    int kt = ks >> 5;
#pragma unroll
    for (int c = 0; c < 4; ++c) {
      v16h bf = load_b_packed(Bp, kt, wave * 4 + c, D / 16, lane);
      acc[c] = wmma16(af, bf, acc[c]);
    }
  }
#pragma unroll
  for (int c = 0; c < 4; ++c) {
    int colT = (wave * 4 + c) * 16;
#pragma unroll
    for (int j = 0; j < 8; ++j) {
      int m = j + ((lane >> 4) << 3);
      int n = (lane & 15) + colT;
      int rr = row0 + m;
      if (rr < nrows) Cout[(size_t)rr * D + n] = acc[c][j];
    }
  }
}

// ---------------- combine: h = relu(conf*strong + (1-conf)*weak) -----------

__global__ void k_combine(const float* __restrict__ strong, const float* __restrict__ weak,
                          const float* __restrict__ conf, float* __restrict__ hout, int n) {
  int i = blockIdx.x * blockDim.x + threadIdx.x;
  if (i >= n) return;
  float c = conf[i >> 8];
  float v = c * strong[i] + (1.0f - c) * weak[i];
  hout[i] = v > 0.0f ? v : 0.0f;
}

// ---------------- output GEMM: out[n,128] = H16[n,256] @ Wp[256,128] -------

__global__ void __launch_bounds__(64) k_out_gemm(
    const half_t* __restrict__ H16, const half_t* __restrict__ Wp,
    float* __restrict__ out, int nrows) {
  __shared__ half_t ldsA[16 * D];  // 8 KB
  int row0 = blockIdx.x * 16;
  int tid = threadIdx.x, wave = tid >> 5, lane = tid & 31;
  for (int i = tid; i < 16 * D; i += 64) {
    int m = i >> 8, k = i & (D - 1);
    int rr = row0 + m;
    ldsA[i] = (rr < nrows) ? H16[(size_t)rr * D + k] : (half_t)0.0f;
  }
  __syncthreads();
  v8f zero = {};
  v8f acc[4] = {zero, zero, zero, zero};
  for (int ks = 0; ks < D; ks += 32) {
    v16h af = load_a_frag(ldsA + ks, D, lane);
    int kt = ks >> 5;
#pragma unroll
    for (int c = 0; c < 4; ++c) {
      v16h bf = load_b_packed(Wp, kt, wave * 4 + c, OUTD / 16, lane);
      acc[c] = wmma16(af, bf, acc[c]);
    }
  }
#pragma unroll
  for (int c = 0; c < 4; ++c) {
    int colT = (wave * 4 + c) * 16;
#pragma unroll
    for (int j = 0; j < 8; ++j) {
      int m = j + ((lane >> 4) << 3);
      int n = (lane & 15) + colT;
      int rr = row0 + m;
      if (rr < nrows) out[(size_t)rr * OUTD + n] = acc[c][j];
    }
  }
}

// ---------------- host launch ----------------------------------------------

extern "C" void kernel_launch(void* const* d_in, const int* in_sizes, int n_in,
                              void* d_out, int out_size, void* d_ws, size_t ws_size,
                              hipStream_t stream) {
  (void)in_sizes; (void)n_in; (void)out_size; (void)ws_size;
  const float* x    = (const float*)d_in[0];
  const int*   ei   = (const int*)d_in[1];
  const float* Wg   = (const float*)d_in[2];
  const float* We   = (const float*)d_in[3];
  const float* Ww   = (const float*)d_in[4];
  const float* W1   = (const float*)d_in[5];
  const float* b1   = (const float*)d_in[6];
  const float* W2   = (const float*)d_in[7];
  const float* b2   = (const float*)d_in[8];
  const float* wc   = (const float*)d_in[9];
  const float* Wout = (const float*)d_in[10];
  const int* src = ei;
  const int* dst = ei + N_EDGES;

  // workspace carve-out
  char* w = (char*)d_ws;
  auto carve = [&](size_t bytes) {
    void* p = (void*)w;
    w += (bytes + 255) & ~(size_t)255;
    return p;
  };
  float*  deg      = (float*)carve((size_t)N_NODES * 4);
  float*  agg      = (float*)carve((size_t)N_NODES * D * 4);
  float*  hA       = (float*)carve((size_t)N_NODES * D * 4);
  float*  hB       = (float*)carve((size_t)N_NODES * D * 4);
  float*  strong   = (float*)carve((size_t)N_NODES * D * 4);
  float*  weak     = (float*)carve((size_t)N_NODES * D * 4);
  half_t* r16      = (half_t*)carve((size_t)N_NODES * R * 2);
  float*  conf     = (float*)carve((size_t)N_NODES * 4);
  int*    topi     = (int*)carve((size_t)N_NODES * 2 * 4);
  float*  topg     = (float*)carve((size_t)N_NODES * 2 * 4);
  int*    counts   = (int*)carve(NE * 4);
  int*    offsets  = (int*)carve((NE + 1) * 4);
  int*    cursors  = (int*)carve(NE * 4);
  int*    asg_node = (int*)carve((size_t)N_NODES * 2 * 4);
  float*  asg_gate = (float*)carve((size_t)N_NODES * 2 * 4);
  half_t* W1p      = (half_t*)carve((size_t)NE * R * D * 2);
  half_t* W2p      = (half_t*)carve((size_t)NE * D * D * 2);
  half_t* Wwp      = (half_t*)carve((size_t)R * D * 2);
  half_t* Woutp    = (half_t*)carve((size_t)D * OUTD * 2);

  const int TB = 256;
  // degrees (constant across layers)
  k_zero_f32<<<(N_NODES + TB - 1) / TB, TB, 0, stream>>>(deg, N_NODES);
  k_degree<<<(N_EDGES + TB - 1) / TB, TB, 0, stream>>>(dst, deg);

  const float* hcur = x;
  float* houts[NL] = {hA, hB, hA};
  const int ND = N_NODES * D;

  for (int l = 0; l < NL; ++l) {
    // aggregation
    k_zero_f32<<<(ND + TB - 1) / TB, TB, 0, stream>>>(agg, ND);
    k_scatter<<<(N_EDGES * 32 + TB - 1) / TB, TB, 0, stream>>>(hcur, src, dst, agg);
    // per-layer weight packing (f32 -> fragment-ready f16, L2-resident)
    k_pack_b<<<(NE * R * D + TB - 1) / TB, TB, 0, stream>>>(W1 + (size_t)l * NE * R * D, W1p, R, D, NE);
    k_pack_b<<<(NE * D * D + TB - 1) / TB, TB, 0, stream>>>(W2 + (size_t)l * NE * D * D, W2p, D, D, NE);
    k_pack_b<<<(R * D + TB - 1) / TB, TB, 0, stream>>>(Ww + (size_t)l * R * D, Wwp, R, D, 1);
    // routing
    k_zero_i32<<<1, 32, 0, stream>>>(counts, NE);
    k_router<<<(N_NODES * 32 + TB - 1) / TB, TB, 0, stream>>>(
        hcur, agg, deg, Wg + (size_t)l * R * NG, We + (size_t)l * R * NE,
        wc + (size_t)l * R, r16, conf, topi, topg, counts);
    k_offsets<<<1, 1, 0, stream>>>(counts, offsets, cursors);
    k_zero_f32<<<(ND + TB - 1) / TB, TB, 0, stream>>>(strong, ND);
    k_assign<<<(N_NODES + TB - 1) / TB, TB, 0, stream>>>(topi, topg, cursors, asg_node, asg_gate);
    // top-2 expert compute (WMMA)
    k_expert<<<(2 * N_NODES) / 16 + NE, 128, 0, stream>>>(
        r16, W1p, W2p, b1 + (size_t)l * NE * D, b2 + (size_t)l * NE * D,
        offsets, asg_node, asg_gate, strong);
    // weak shared expert (WMMA)
    k_weak_gemm<<<(N_NODES + 15) / 16, 128, 0, stream>>>(r16, Wwp, weak, N_NODES);
    // blend
    k_combine<<<(ND + TB - 1) / TB, TB, 0, stream>>>(strong, weak, conf, houts[l], ND);
    hcur = houts[l];
  }

  // final projection (WMMA); reuse r16 as h16 staging
  k_f32_to_f16<<<(ND + TB - 1) / TB, TB, 0, stream>>>(hcur, r16, ND);
  k_pack_b<<<(D * OUTD + TB - 1) / TB, TB, 0, stream>>>(Wout, Woutp, D, OUTD, 1);
  k_out_gemm<<<(N_NODES + 15) / 16, 64, 0, stream>>>(r16, Woutp, (float*)d_out, N_NODES);
}
